// EMALayer_72060961292386
// MI455X (gfx1250) — matrix operand order repaired and checked by reading, not compiled, not run on hardware
//
#include <hip/hip_runtime.h>
#include <hip/hip_bf16.h>

// ---------------------------------------------------------------------------
// EMA layer for MI455X (gfx1250, wave32, WMMA), split-bf16 (bf16x3) GEMMs.
//   u = silu(x @ W_in^T + b_in)            GEMM1 -> u hi/lo bf16 planes
//   h[t] = (1-g) h[t-1] + g u[t]           chunked scan, g = sigmoid(weight)
//   y = tanh(h @ W_out^T + b_out)          GEMM2 -> f32 output
// f32 operands are pre-split ONCE into (hi, lo) bf16 planes; the GEMM inner
// loop is pure global_load_b128 + v_wmma.  64x32 tile per wave: 24 loads for
// 24 WMMAs per K=32 step.
// ---------------------------------------------------------------------------

typedef __attribute__((ext_vector_type(16))) __bf16 v16bf;
typedef __attribute__((ext_vector_type(8)))  float  v8f;

#define T_DIM 4096
#define B_DIM 4
#define D_DIM 1024
#define M_DIM (T_DIM * B_DIM)      // 16384 rows for both GEMMs
#define BD    (B_DIM * D_DIM)      // 4096 scan channels
#define NCHUNK 32
#define CHUNKLEN (T_DIM / NCHUNK)  // 128

#define MT 64                      // wave tile rows (4 x 16)
#define NT 32                      // wave tile cols (2 x 16)

// ---------------------------------------------------------------------------
__device__ __forceinline__ v8f wmma_bf16(v16bf a, v16bf b, v8f c) {
  return __builtin_amdgcn_wmma_f32_16x16x32_bf16(
      /*neg_a=*/false, a, /*neg_b=*/false, b,
      /*c_mod=*/(short)0, c, /*reuse_a=*/false, /*reuse_b=*/false);
}

union FragU {
  v16bf v;
  float4 q[2];
};

// A fragment (16x32 bf16): p = &plane[row*K + k0 + half*8].
// halves [0..7] = K k0+half*8 .. +7 ; halves [8..15] = +16 further.
__device__ __forceinline__ v16bf loadFragA(const __bf16* p) {
  FragU f;
  f.q[0] = *(const float4*)(p);
  f.q[1] = *(const float4*)(p + 16);
  return f.v;
}

// B fragment (32x16 bf16): p = &plane[col*K + k0 + half*8]; [0..7] and +16 further.
__device__ __forceinline__ v16bf loadFragB(const __bf16* p) {
  FragU f;
  f.q[0] = *(const float4*)(p);
  f.q[1] = *(const float4*)(p + 16);
  return f.v;
}

struct alignas(8) BF4 { __bf16 a, b, c, d; };

// ---------------------------------------------------------------------------
// One-time f32 -> (hi, lo) bf16 split.  v ~= hi + lo.
// ---------------------------------------------------------------------------
__global__ void __launch_bounds__(256)
split_f32_bf16(const float* __restrict__ src, __bf16* __restrict__ hi,
               __bf16* __restrict__ lo, int n4) {
  const int i = blockIdx.x * blockDim.x + threadIdx.x;
  if (i >= n4) return;
  const float4 v = ((const float4*)src)[i];
  const __bf16 h0 = (__bf16)v.x, h1 = (__bf16)v.y;
  const __bf16 h2 = (__bf16)v.z, h3 = (__bf16)v.w;
  BF4 H = {h0, h1, h2, h3};
  BF4 L = {(__bf16)(v.x - (float)h0), (__bf16)(v.y - (float)h1),
           (__bf16)(v.z - (float)h2), (__bf16)(v.w - (float)h3)};
  ((BF4*)hi)[i] = H;
  ((BF4*)lo)[i] = L;
}

// ---------------------------------------------------------------------------
// GEMM on pre-split planes: C[m,n] = act( sum_k A[m,k]*W[n,k] + bias[n] )
// using hi*hi + hi*lo + lo*hi.  One wave per 64x32 output tile.
// ACT 0: SiLU, emit hi/lo bf16 planes.  ACT 1: tanh, emit f32.
// ---------------------------------------------------------------------------
template <int ACT>
__global__ void __launch_bounds__(256)
gemm_planes(const __bf16* __restrict__ Ahi, const __bf16* __restrict__ Alo,
            const __bf16* __restrict__ Bhi, const __bf16* __restrict__ Blo,
            const float* __restrict__ bias,
            __bf16* __restrict__ Chi, __bf16* __restrict__ Clo,
            float* __restrict__ Cf,
            int M, int N, int K) {
  const int wid  = (blockIdx.x * blockDim.x + threadIdx.x) >> 5;
  const int lane = threadIdx.x & 31;
  const int tilesN = N / NT;
  const int m0 = (wid / tilesN) * MT;
  const int n0 = (wid % tilesN) * NT;
  if (m0 >= M) return;

  const int half = lane >> 4;
  const int l16  = lane & 15;

  size_t aoff[4];
#pragma unroll
  for (int mi = 0; mi < 4; ++mi)
    aoff[mi] = (size_t)(m0 + mi * 16 + l16) * K + half * 8;
  const size_t boff0 = (size_t)(n0 + l16) * K + half * 8;
  const size_t boff1 = (size_t)(n0 + 16 + l16) * K + half * 8;

  v8f acc[4][2] = {};

#pragma unroll 2
  for (int k0 = 0; k0 < K; k0 += 32) {
    // unconditional speculative prefetch of the A stream two blocks ahead
    __builtin_prefetch(Ahi + aoff[0] + k0 + 64, 0, 1);
    __builtin_prefetch(Ahi + aoff[2] + k0 + 64, 0, 1);

    v16bf bh[2], bl[2];
    bh[0] = loadFragB(Bhi + boff0 + k0);
    bl[0] = loadFragB(Blo + boff0 + k0);
    bh[1] = loadFragB(Bhi + boff1 + k0);
    bl[1] = loadFragB(Blo + boff1 + k0);
#pragma unroll
    for (int mi = 0; mi < 4; ++mi) {
      const v16bf ah = loadFragA(Ahi + aoff[mi] + k0);
      const v16bf al = loadFragA(Alo + aoff[mi] + k0);
#pragma unroll
      for (int ni = 0; ni < 2; ++ni) {
        acc[mi][ni] = wmma_bf16(ah, bh[ni], acc[mi][ni]);
        acc[mi][ni] = wmma_bf16(ah, bl[ni], acc[mi][ni]);
        acc[mi][ni] = wmma_bf16(al, bh[ni], acc[mi][ni]);
      }
    }
  }

  // Epilogue.  C/D layout: lane<16 -> rows j (VGPR j), col = lane;
  // lane>=16 -> rows 8+j, col = lane-16.
#pragma unroll
  for (int ni = 0; ni < 2; ++ni) {
    const int col = n0 + ni * 16 + l16;
    const float bv = bias[col];
#pragma unroll
    for (int mi = 0; mi < 4; ++mi) {
#pragma unroll
      for (int j = 0; j < 8; ++j) {
        const int row = m0 + mi * 16 + half * 8 + j;
        float v = acc[mi][ni][j] + bv;
        if (ACT == 0) {
          v = v / (1.f + __expf(-v));       // SiLU
          const __bf16 h = (__bf16)v;
          Chi[(size_t)row * N + col] = h;
          Clo[(size_t)row * N + col] = (__bf16)(v - (float)h);
        } else {
          Cf[(size_t)row * N + col] = tanhf(v);
        }
      }
    }
  }
}

// ---------------------------------------------------------------------------
// EMA scan over hi/lo planes.  u layout: [T][B*D], channel = b*D + d fast.
// ---------------------------------------------------------------------------
__global__ void __launch_bounds__(256)
ema_chunk_ends(const __bf16* __restrict__ uhi, const __bf16* __restrict__ ulo,
               const float* __restrict__ weight, float* __restrict__ carry) {
  const int tid = blockIdx.x * blockDim.x + threadIdx.x;
  const int ch = tid & (BD - 1);
  const int c  = tid >> 12;                 // / BD
  if (c >= NCHUNK) return;
  const int d = ch & (D_DIM - 1);
  const float g = 1.f / (1.f + __expf(-weight[d]));
  const float a = 1.f - g;
  const size_t base = (size_t)c * CHUNKLEN * BD + ch;
  float h = 0.f;
#pragma unroll 4
  for (int i = 0; i < CHUNKLEN; ++i) {
    const size_t idx = base + (size_t)i * BD;
    const float uv = (float)uhi[idx] + (float)ulo[idx];
    h = fmaf(a, h, g * uv);
  }
  carry[c * BD + ch] = h;
}

__global__ void __launch_bounds__(256)
ema_combine(const float* __restrict__ weight, const float* __restrict__ carry,
            float* __restrict__ Hc) {
  const int ch = blockIdx.x * blockDim.x + threadIdx.x;
  if (ch >= BD) return;
  const int d = ch & (D_DIM - 1);
  const float g = 1.f / (1.f + __expf(-weight[d]));
  const float a = 1.f - g;
  const float aL = __powf(a, (float)CHUNKLEN);
  float S = 0.f;
#pragma unroll
  for (int c = 0; c < NCHUNK; ++c) {
    Hc[c * BD + ch] = S;
    S = fmaf(aL, S, carry[c * BD + ch]);
  }
}

__global__ void __launch_bounds__(256)
ema_apply(const __bf16* __restrict__ uhi, const __bf16* __restrict__ ulo,
          const float* __restrict__ weight, const float* __restrict__ Hc,
          __bf16* __restrict__ vhi, __bf16* __restrict__ vlo) {
  const int tid = blockIdx.x * blockDim.x + threadIdx.x;
  const int ch = tid & (BD - 1);
  const int c  = tid >> 12;
  if (c >= NCHUNK) return;
  const int d = ch & (D_DIM - 1);
  const float g = 1.f / (1.f + __expf(-weight[d]));
  const float a = 1.f - g;
  const size_t base = (size_t)c * CHUNKLEN * BD + ch;
  float h = Hc[c * BD + ch];
#pragma unroll 4
  for (int i = 0; i < CHUNKLEN; ++i) {
    const size_t idx = base + (size_t)i * BD;
    const float uv = (float)uhi[idx] + (float)ulo[idx];
    h = fmaf(a, h, g * uv);
    const __bf16 hh = (__bf16)h;
    vhi[idx] = hh;
    vlo[idx] = (__bf16)(h - (float)hh);
  }
}

// ---------------------------------------------------------------------------
extern "C" void kernel_launch(void* const* d_in, const int* in_sizes, int n_in,
                              void* d_out, int out_size, void* d_ws, size_t ws_size,
                              hipStream_t stream) {
  const float* x      = (const float*)d_in[0];
  const float* W_in   = (const float*)d_in[1];
  const float* b_in   = (const float*)d_in[2];
  const float* W_out  = (const float*)d_in[3];
  const float* b_out  = (const float*)d_in[4];
  const float* weight = (const float*)d_in[5];
  float* out = (float*)d_out;

  char* ws = (char*)d_ws;
  const size_t planeB = (size_t)M_DIM * D_DIM * sizeof(__bf16);   // 32 MB
  const size_t wB     = (size_t)D_DIM * D_DIM * sizeof(__bf16);   //  2 MB

  // x planes are dead after GEMM1; v planes alias them.
  __bf16* x_hi = (__bf16*)(ws);
  __bf16* x_lo = (__bf16*)(ws + planeB);
  __bf16* v_hi = x_hi;
  __bf16* v_lo = x_lo;
  __bf16* u_hi = (__bf16*)(ws + 2 * planeB);
  __bf16* u_lo = (__bf16*)(ws + 3 * planeB);
  __bf16* wi_hi = (__bf16*)(ws + 4 * planeB);
  __bf16* wi_lo = (__bf16*)(ws + 4 * planeB + wB);
  __bf16* wo_hi = (__bf16*)(ws + 4 * planeB + 2 * wB);
  __bf16* wo_lo = (__bf16*)(ws + 4 * planeB + 3 * wB);
  float*  carry = (float*)(ws + 4 * planeB + 4 * wB);
  float*  Hc    = carry + NCHUNK * BD;

  dim3 block(256);

  // 0) one-time f32 -> bf16 hi/lo splits
  {
    const int nx4 = (M_DIM * D_DIM) / 4;
    const int nw4 = (D_DIM * D_DIM) / 4;
    split_f32_bf16<<<dim3(nx4 / 256), block, 0, stream>>>(x, x_hi, x_lo, nx4);
    split_f32_bf16<<<dim3(nw4 / 256), block, 0, stream>>>(W_in, wi_hi, wi_lo, nw4);
    split_f32_bf16<<<dim3(nw4 / 256), block, 0, stream>>>(W_out, wo_hi, wo_lo, nw4);
  }

  const int waves  = (M_DIM / MT) * (D_DIM / NT);   // 8192 tiles
  const int blocks = waves / 8;                      // 8 waves / 256-thr block

  // 1) u = silu(x @ W_in^T + b_in)  -> u hi/lo planes
  gemm_planes<0><<<dim3(blocks), block, 0, stream>>>(
      x_hi, x_lo, wi_hi, wi_lo, b_in, u_hi, u_lo, nullptr,
      M_DIM, D_DIM, D_DIM);

  // 2) chunked EMA scan (u planes -> v planes, aliasing dead x planes)
  ema_chunk_ends<<<dim3((BD * NCHUNK) / 256), block, 0, stream>>>(
      u_hi, u_lo, weight, carry);
  ema_combine<<<dim3(BD / 256), block, 0, stream>>>(weight, carry, Hc);
  ema_apply<<<dim3((BD * NCHUNK) / 256), block, 0, stream>>>(
      u_hi, u_lo, weight, Hc, v_hi, v_lo);

  // 3) y = tanh(v @ W_out^T + b_out) -> f32 output
  gemm_planes<1><<<dim3(blocks), block, 0, stream>>>(
      v_hi, v_lo, wo_hi, wo_lo, b_out, nullptr, nullptr, out,
      M_DIM, D_DIM, D_DIM);
}
